// EnhancedPhysicsGNN_29643864277620
// MI455X (gfx1250) — compile-verified, matrix operand order checked by reading, not run on hardware
//
#include <hip/hip_runtime.h>
#include <math.h>

// ---------------------------------------------------------------------------
// EnhancedPhysicsGNN forward for MI455X (gfx1250, wave32, WMMA).
// Dense matmuls use v_wmma_f32_16x16x32_f16 (f16 in, f32 accumulate),
// double-buffered LDS, 8 WMMA per wave per K-step, 1 barrier per K-step.
// Edge scatter/segment ops use f32 global atomics (bandwidth bound at 23 TB/s).
// ---------------------------------------------------------------------------

#define GN  10000      // nodes
#define GE  160000     // edges
#define GB  64         // graphs
#define GF  8          // node features
#define GDE 4          // edge features
#define GH  256        // hidden
#define GHEADS 4
#define GDH 64
#define GLAYERS 8
#define GSTEPS 4

typedef __attribute__((ext_vector_type(16))) _Float16 v16h;
typedef __attribute__((ext_vector_type(8)))  float    v8f;
typedef __attribute__((ext_vector_type(4)))  float    f32x4;
typedef __attribute__((ext_vector_type(4)))  unsigned int u32x4;

__device__ __forceinline__ float sigmf(float x) { return 1.0f / (1.0f + __expf(-x)); }
__device__ __forceinline__ float siluf(float x) { return x * sigmf(x); }

__device__ __forceinline__ float atomicMaxF(float* addr, float val) {
  int old = __float_as_int(*addr);
  while (val > __int_as_float(old)) {
    int assumed = old;
    old = atomicCAS((int*)addr, assumed, __float_as_int(val));
    if (old == assumed) break;
  }
  return __int_as_float(old);
}

__device__ __forceinline__ float blockReduceSum(float v, float* red, int n) {
  int t = threadIdx.x;
  red[t] = v;
  __syncthreads();
  for (int s = n >> 1; s > 0; s >>= 1) {
    if (t < s) red[t] += red[t + s];
    __syncthreads();
  }
  float r = red[0];
  __syncthreads();
  return r;
}

// ---------------------------------------------------------------------------
// WMMA GEMM:  C[M x Nc] = A[M x K] @ W[K x Nc] + bias
// Requirements: K % 32 == 0, Nc % 128 == 0 (M may be ragged).
// Block = 128 threads (4 waves), tile 64x128, BK = 32, double-buffered LDS.
// Each wave owns a 16-row strip -> 8 x v_wmma_f32_16x16x32_f16 per K-step,
// one s_barrier per K-step; next K-tile staged in registers during compute;
// global_prefetch_b8 covers the tile after next. B-tile transpose staging
// packs (k,k+1) pairs -> single b32 LDS stores (v_cvt_pk_f16_f32).
// ---------------------------------------------------------------------------
#define TBM 64
#define TBN 128
#define TBK 32

__global__ __launch_bounds__(128) void k_gemm_f16(
    const float* __restrict__ A, const float* __restrict__ W,
    const float* __restrict__ bias, float* __restrict__ C,
    int M, int K, int Nc)
{
  __shared__ __align__(16) _Float16 As[2][TBM][TBK];   // 2 x 4 KB
  __shared__ __align__(16) _Float16 Bs[2][TBN][TBK];   // 2 x 8 KB

  const int tid  = threadIdx.x;
  const int wave = tid >> 5;
  const int lane = tid & 31;
  const int n0   = blockIdx.x * TBN;
  const int m0   = blockIdx.y * TBM;
  const bool fullM = (m0 + TBM) <= M;

  const v8f z8 = {0.f, 0.f, 0.f, 0.f, 0.f, 0.f, 0.f, 0.f};
  v8f acc[8];
#pragma unroll
  for (int i = 0; i < 8; i++) acc[i] = z8;

  // A staging map: 64x32 f32 = 128 thr x 4 f32x4
  const int arow = tid >> 3;          // 0..15 (row base, step 16)
  const int acol = (tid & 7) * 4;     // 0..28
  // B staging map: 32x128 f32; thread covers 4 (k-pair, n-group) tasks:
  // kpair = (tid>>5) + p*4 (p=0..3)  -> k = 2*kpair, 2*kpair+1
  // n      = (tid&31)*4 .. +3
  const int kpb = tid >> 5;           // 0..3
  const int wn  = (tid & 31) * 4;     // 0..124

  f32x4 areg[4], breg0[4], breg1[4];

  auto loadTiles = [&](int k0) {
#pragma unroll
    for (int r = 0; r < 4; r++) {
      int gm = m0 + arow + r * 16;
      f32x4 z = {0.f, 0.f, 0.f, 0.f};
      areg[r] = z;
      if (fullM || gm < M)
        areg[r] = *(const f32x4*)(A + (size_t)gm * K + k0 + acol);
    }
#pragma unroll
    for (int p = 0; p < 4; p++) {
      int kk = (kpb + p * 4) * 2;
      breg0[p] = *(const f32x4*)(W + (size_t)(k0 + kk) * Nc + n0 + wn);
      breg1[p] = *(const f32x4*)(W + (size_t)(k0 + kk + 1) * Nc + n0 + wn);
    }
  };
  auto storeTiles = [&](int buf) {
#pragma unroll
    for (int r = 0; r < 4; r++) {
      int row = arow + r * 16;
      As[buf][row][acol + 0] = (_Float16)areg[r][0];
      As[buf][row][acol + 1] = (_Float16)areg[r][1];
      As[buf][row][acol + 2] = (_Float16)areg[r][2];
      As[buf][row][acol + 3] = (_Float16)areg[r][3];
    }
#pragma unroll
    for (int p = 0; p < 4; p++) {
      int kk = (kpb + p * 4) * 2;     // even -> 4-byte aligned in [n][k] layout
#pragma unroll
      for (int j = 0; j < 4; j++) {
        union { unsigned int u; _Float16 h[2]; } pk;
        pk.h[0] = (_Float16)breg0[p][j];   // k
        pk.h[1] = (_Float16)breg1[p][j];   // k+1
        *(unsigned int*)&Bs[buf][wn + j][kk] = pk.u;
      }
    }
  };

  const int half = lane >> 4;         // 0/1
  const int l16  = lane & 15;
  const int kb   = half * 8;          // K sub-offset per documented f16 layout

  loadTiles(0);
  storeTiles(0);
  __syncthreads();

  int buf = 0;
  for (int k0 = 0; k0 < K; k0 += TBK) {
    const bool hasNext = (k0 + TBK) < K;
    if (hasNext) loadTiles(k0 + TBK);          // overlap with compute below
    if (k0 + 2 * TBK < K) {                    // gfx1250 global_prefetch_b8
      __builtin_prefetch(A + (size_t)(m0 + arow) * K + k0 + 2 * TBK + acol, 0, 1);
      __builtin_prefetch(W + (size_t)(k0 + 2 * TBK + kpb * 2) * Nc + n0 + wn, 0, 1);
    }

    // A fragment: this wave's 16-row strip
    union { u32x4 q[2]; v16h v; } af;
    const _Float16* ap = &As[buf][wave * 16 + l16][0];
    af.q[0] = *(const u32x4*)(ap + kb);
    af.q[1] = *(const u32x4*)(ap + kb + 16);

#pragma unroll
    for (int nt = 0; nt < 8; nt++) {
      union { u32x4 q[2]; v16h v; } bf;
      const _Float16* bp = &Bs[buf][nt * 16 + l16][0];
      bf.q[0] = *(const u32x4*)(bp + kb);
      bf.q[1] = *(const u32x4*)(bp + kb + 16);
      acc[nt] = __builtin_amdgcn_wmma_f32_16x16x32_f16(
          false, af.v, false, bf.v, (short)0, acc[nt], false, false);
    }

    if (hasNext) {
      buf ^= 1;
      storeTiles(buf);
      __syncthreads();
    }
  }

  // epilogue: C/D layout -> lanes 0-15: M=r, lanes 16-31: M=r+8; N = lane&15
#pragma unroll
  for (int nt = 0; nt < 8; nt++) {
    int n = n0 + nt * 16 + l16;
    float bv = bias ? bias[n] : 0.f;
#pragma unroll
    for (int r = 0; r < 8; r++) {
      int m = m0 + wave * 16 + half * 8 + r;
      if (fullM || m < M) C[(size_t)m * Nc + n] = acc[nt][r] + bv;
    }
  }
}

// ---------------------------------------------------------------------------
// Small K GEMM (K <= 16): one block per row
// ---------------------------------------------------------------------------
__global__ void k_small_gemm(const float* __restrict__ A, const float* __restrict__ W,
                             const float* __restrict__ bias, float* __restrict__ C,
                             int K, int Nc)
{
  int row = blockIdx.x, j = threadIdx.x;
  float acc = bias[j];
  for (int k = 0; k < K; k++) acc += A[(size_t)row * K + k] * W[k * Nc + j];
  C[(size_t)row * Nc + j] = acc;
}

// LayerNorm(+affine) then SiLU over one row; blockDim.x == Hd
__global__ void k_ln_silu(const float* __restrict__ in, const float* __restrict__ g,
                          const float* __restrict__ b, float* __restrict__ out, int Hd)
{
  __shared__ float red[256];
  int row = blockIdx.x, j = threadIdx.x;
  float x = in[(size_t)row * Hd + j];
  float mu = blockReduceSum(x, red, Hd) / (float)Hd;
  float d = x - mu;
  float var = blockReduceSum(d * d, red, Hd) / (float)Hd;
  float y = d * rsqrtf(var + 1e-5f) * g[j] + b[j];
  out[(size_t)row * Hd + j] = siluf(y);
}

// BatchNorm stats over rows (per column)
__global__ void k_bn_stats(const float* __restrict__ x, float* __restrict__ mu,
                           float* __restrict__ var, int M, int Hd)
{
  __shared__ float r1[256], r2[256];
  int j = blockIdx.x, t = threadIdx.x;
  float s = 0.f, s2 = 0.f;
  for (int i = t; i < M; i += blockDim.x) {
    float v = x[(size_t)i * Hd + j];
    s += v; s2 += v * v;
  }
  r1[t] = s; r2[t] = s2;
  __syncthreads();
  for (int st = blockDim.x >> 1; st > 0; st >>= 1) {
    if (t < st) { r1[t] += r1[t + st]; r2[t] += r2[t + st]; }
    __syncthreads();
  }
  if (t == 0) {
    float m = r1[0] / (float)M;
    mu[j] = m;
    var[j] = r2[0] / (float)M - m * m;
  }
}

// h += bn(hn)
__global__ void k_bn_residual(float* __restrict__ h, const float* __restrict__ hn,
                              const float* __restrict__ mu, const float* __restrict__ var,
                              const float* __restrict__ g, const float* __restrict__ b,
                              size_t total, int Hd)
{
  size_t idx = (size_t)blockIdx.x * blockDim.x + threadIdx.x;
  if (idx >= total) return;
  int j = (int)(idx % Hd);
  h[idx] += (hn[idx] - mu[j]) * rsqrtf(var[j] + 1e-5f) * g[j] + b[j];
}

__global__ void k_fill(float* __restrict__ p, float v, size_t n) {
  size_t i = (size_t)blockIdx.x * blockDim.x + threadIdx.x;
  if (i < n) p[i] = v;
}

__global__ void k_add(const float* __restrict__ a, const float* __restrict__ b,
                      float* __restrict__ c, size_t n) {
  size_t i = (size_t)blockIdx.x * blockDim.x + threadIdx.x;
  if (i < n) c[i] = a[i] + b[i];
}

// even layer: aggr[dst] += relu(h[src] + edge_attr@We + bee)
__global__ void k_edge_aggr(const float* __restrict__ h, const float* __restrict__ ea,
                            const float* __restrict__ We, const float* __restrict__ bee,
                            const int* __restrict__ src, const int* __restrict__ dst,
                            float* __restrict__ aggr)
{
  int e = blockIdx.x, j = threadIdx.x;
  int s = src[e], d = dst[e];
  float ew = bee[j];
#pragma unroll
  for (int f = 0; f < GDE; f++) ew += ea[(size_t)e * GDE + f] * We[f * GH + j];
  float m = h[(size_t)s * GH + j] + ew;
  m = m > 0.f ? m : 0.f;
  atomicAdd(&aggr[(size_t)d * GH + j], m);
}

// attention logits per (edge, head) + running per-(node,head) max
__global__ void k_attn_logits(const float* __restrict__ Q, const float* __restrict__ Km,
                              const float* __restrict__ ea, const float* __restrict__ We,
                              const float* __restrict__ bee,
                              const int* __restrict__ src, const int* __restrict__ dst,
                              float* __restrict__ logits, float* __restrict__ nmax)
{
  __shared__ float red[256];
  int e = blockIdx.x, j = threadIdx.x;
  int s = src[e], d = dst[e];
  float ew = bee[j];
#pragma unroll
  for (int f = 0; f < GDE; f++) ew += ea[(size_t)e * GDE + f] * We[f * GH + j];
  red[j] = Q[(size_t)d * GH + j] * (Km[(size_t)s * GH + j] + ew);
  __syncthreads();
  for (int st = 32; st > 0; st >>= 1) {        // reduce within each head (64 lanes)
    if ((j & 63) < st) red[j] += red[j + st];
    __syncthreads();
  }
  if ((j & 63) == 0) {
    int hh = j >> 6;
    float lg = red[j] * 0.125f;                // / sqrt(DH=64)
    logits[(size_t)e * GHEADS + hh] = lg;
    atomicMaxF(&nmax[(size_t)d * GHEADS + hh], lg);
  }
}

__global__ void k_attn_expsum(const float* __restrict__ logits, const float* __restrict__ nmax,
                              const int* __restrict__ dst, float* __restrict__ alpha,
                              float* __restrict__ nsum, int nE)
{
  int idx = blockIdx.x * blockDim.x + threadIdx.x;
  if (idx >= nE * GHEADS) return;
  int e = idx >> 2, hh = idx & 3;
  int d = dst[e];
  float a = __expf(logits[idx] - nmax[(size_t)d * GHEADS + hh]);
  alpha[idx] = a;
  atomicAdd(&nsum[(size_t)d * GHEADS + hh], a);
}

__global__ void k_attn_out(const float* __restrict__ V, const float* __restrict__ ea,
                           const float* __restrict__ We, const float* __restrict__ bee,
                           const int* __restrict__ src, const int* __restrict__ dst,
                           const float* __restrict__ alpha, const float* __restrict__ nsum,
                           float* __restrict__ out)
{
  int e = blockIdx.x, j = threadIdx.x;
  int s = src[e], d = dst[e];
  int hh = j >> 6;
  float w = alpha[(size_t)e * GHEADS + hh] / (nsum[(size_t)d * GHEADS + hh] + 1e-16f);
  float ew = bee[j];
#pragma unroll
  for (int f = 0; f < GDE; f++) ew += ea[(size_t)e * GDE + f] * We[f * GH + j];
  atomicAdd(&out[(size_t)d * GH + j], w * (V[(size_t)s * GH + j] + ew));
}

// beta = sigmoid([out, r, out-r] @ Wbeta); hn = beta*r + (1-beta)*out
__global__ void k_beta(const float* __restrict__ out, const float* __restrict__ R,
                       const float* __restrict__ Wb, float* __restrict__ hn)
{
  __shared__ float red[256];
  int n = blockIdx.x, j = threadIdx.x;
  float o = out[(size_t)n * GH + j];
  float r = R[(size_t)n * GH + j];
  float part = o * Wb[j] + r * Wb[GH + j] + (o - r) * Wb[2 * GH + j];
  float tot = blockReduceSum(part, red, GH);
  float beta = sigmf(tot);
  hn[(size_t)n * GH + j] = beta * r + (1.f - beta) * o;
}

__global__ void k_lstm(const float* __restrict__ gA, const float* __restrict__ gB,
                       float* __restrict__ cs, float* __restrict__ hs)
{
  int idx = blockIdx.x * blockDim.x + threadIdx.x;
  if (idx >= GB * GH) return;
  int b = idx >> 8, j = idx & 255;
  const float* a = gA + (size_t)b * 4 * GH;
  const float* g = gB + (size_t)b * 4 * GH;
  float ig = a[j] + g[j];
  float fg = a[GH + j] + g[GH + j];
  float gg = a[2 * GH + j] + g[2 * GH + j];
  float og = a[3 * GH + j] + g[3 * GH + j];
  float c = sigmf(fg) * cs[idx] + sigmf(ig) * tanhf(gg);
  cs[idx] = c;
  hs[idx] = sigmf(og) * tanhf(c);
}

// en[n] = h[n] . hs[batch[n]] ; track per-graph max
__global__ void k_en(const float* __restrict__ h, const float* __restrict__ hs,
                     const int* __restrict__ batch, float* __restrict__ en,
                     float* __restrict__ bmax)
{
  __shared__ float red[256];
  int n = blockIdx.x, j = threadIdx.x;
  int b = batch[n];
  float tot = blockReduceSum(h[(size_t)n * GH + j] * hs[(size_t)b * GH + j], red, GH);
  if (j == 0) { en[n] = tot; atomicMaxF(&bmax[b], tot); }
}

__global__ void k_segexp(const float* __restrict__ s, const int* __restrict__ batch,
                         const float* __restrict__ bmax, float* __restrict__ a,
                         float* __restrict__ bsum, int n)
{
  int i = blockIdx.x * blockDim.x + threadIdx.x;
  if (i >= n) return;
  int b = batch[i];
  float v = __expf(s[i] - bmax[b]);
  a[i] = v;
  atomicAdd(&bsum[b], v);
}

__global__ void k_segwsum(const float* __restrict__ a, const float* __restrict__ bsum,
                          const int* __restrict__ batch, const float* __restrict__ h,
                          float* __restrict__ outB)
{
  int n = blockIdx.x, j = threadIdx.x;
  int b = batch[n];
  float w = a[n] / (bsum[b] + 1e-16f);
  atomicAdd(&outB[(size_t)b * GH + j], w * h[(size_t)n * GH + j]);
}

__global__ void k_qstar(const float* __restrict__ hs, const float* __restrict__ rr,
                        float* __restrict__ qst)
{
  int idx = blockIdx.x * blockDim.x + threadIdx.x;
  if (idx >= GB * 2 * GH) return;
  int b = idx >> 9, j = idx & 511;
  qst[idx] = (j < GH) ? hs[(size_t)b * GH + j] : rr[(size_t)b * GH + (j - GH)];
}

// gate[n] = silu(G1[n]) . Wg2 + bg2 ; track per-graph max
__global__ void k_gate(const float* __restrict__ G1, const float* __restrict__ Wg2,
                       const float* __restrict__ bg2, const int* __restrict__ batch,
                       float* __restrict__ gate, float* __restrict__ bmax)
{
  __shared__ float red[256];
  int n = blockIdx.x, j = threadIdx.x;   // 128 threads
  float t = siluf(G1[(size_t)n * 128 + j]);
  float tot = blockReduceSum(t * Wg2[j], red, 128);
  if (j == 0) {
    float g = tot + bg2[0];
    gate[n] = g;
    atomicMaxF(&bmax[batch[n]], g);
  }
}

__global__ void k_gfeat(const float* Om, const float* De, const float* En,
                        const float* ss, const float* tr, const float* rd,
                        const float* ce, const float* nA, const float* nB,
                        float* __restrict__ g10)
{
  int b = threadIdx.x;
  if (b >= GB) return;
  float s = ss[b], c = ce[b];
  float* o = g10 + b * 10;
  o[0] = Om[b]; o[1] = De[b]; o[2] = En[b] / s; o[3] = tr[b] / s; o[4] = rd[b];
  o[5] = s; o[6] = c; o[7] = c / logf(s + 1e-6f); o[8] = nA[b]; o[9] = nB[b];
}

__global__ void k_concat3(const float* __restrict__ qst, const float* __restrict__ ga,
                          const float* __restrict__ gf, float* __restrict__ comb)
{
  int idx = blockIdx.x * blockDim.x + threadIdx.x;
  if (idx >= GB * 4 * GH) return;
  int b = idx >> 10, j = idx & 1023;
  float v;
  if (j < 512)      v = qst[(size_t)b * 512 + j];
  else if (j < 768) v = ga[(size_t)b * GH + (j - 512)];
  else              v = gf[(size_t)b * GH + (j - 768)];
  comb[idx] = v;
}

__global__ void k_final(const float* __restrict__ t2, const float* __restrict__ Wf3,
                        const float* __restrict__ bf3, float* __restrict__ out)
{
  __shared__ float red[256];
  int b = blockIdx.x, j = threadIdx.x;   // 128 threads
  float tot = blockReduceSum(t2[(size_t)b * 128 + j] * Wf3[j], red, 128);
  if (j == 0) out[b] = tot + bf3[0];
}

// ---------------------------------------------------------------------------
// host side
// ---------------------------------------------------------------------------
static inline void gemm(const float* A, const float* W, const float* bias, float* C,
                        int M, int K, int Nc, hipStream_t s)
{
  dim3 g(Nc / TBN, (M + TBM - 1) / TBM);
  k_gemm_f16<<<g, dim3(128), 0, s>>>(A, W, bias, C, M, K, Nc);
}
static inline size_t gup(size_t n) { return (n + 255) / 256; }

extern "C" void kernel_launch(void* const* d_in, const int* in_sizes, int n_in,
                              void* d_out, int out_size, void* d_ws, size_t ws_size,
                              hipStream_t stream)
{
  (void)in_sizes; (void)ws_size; (void)out_size;
  auto P = [&](int i) { return (const float*)d_in[i]; };

  // top-level inputs (setup_inputs order)
  const float* x    = P(0);     // N x F
  const float* ea   = P(1);     // E x DE
  const float* Om   = P(2);
  const float* De   = P(3);
  const float* En   = P(4);
  const float* ssz  = P(5);
  const float* trd  = P(6);
  const float* rdd  = P(7);
  const float* cen  = P(8);
  const float* nA   = P(9);
  const float* nB   = P(10);
  const int* ei     = (const int*)d_in[n_in - 2];  // 2 x E
  const int* batch  = (const int*)d_in[n_in - 1];  // N
  const int* src    = ei;
  const int* dst    = ei + GE;

  // params leaves (sorted pytree key order, starting at index 11):
  // Wf1 Wf2 Wf3 Wg1 Wg2 Wgf1 Wgf2 Whh Wi Wih
  // bef1 bef2 begf bei bf1 bf2 bf3 bg1 bg2 bgf1 bgf2 bhh bi bih
  // gf1 gf2 ggf gi  then layers[0..7]
  const float *Wf1=P(11),*Wf2=P(12),*Wf3=P(13),*Wg1=P(14),*Wg2=P(15),
              *Wgf1=P(16),*Wgf2=P(17),*Whh=P(18),*Wi=P(19),*Wih=P(20),
              *bef1=P(21),*bef2=P(22),*begf=P(23),*bei=P(24),*bf1=P(25),
              *bf2=P(26),*bf3=P(27),*bg1=P(28),*bg2=P(29),*bgf1=P(30),
              *bgf2=P(31),*bhh=P(32),*bi=P(33),*bih=P(34),
              *gf1=P(35),*gf2=P(36),*ggf=P(37),*gi=P(38);
  int pidx = 39;

  // workspace layout (floats)
  float* ws = (float*)d_ws;
  const size_t NH = (size_t)GN * GH;
  float* h    = ws;
  float* b1p  = ws + NH;
  float* b2p  = ws + 2 * NH;
  float* b3p  = ws + 3 * NH;
  float* b4p  = ws + 4 * NH;
  float* b5p  = ws + 5 * NH;
  float* eL   = ws + 6 * NH;                 // E*4
  float* eA   = eL + (size_t)GE * 4;         // E*4
  float* nM   = eA + (size_t)GE * 4;         // N*4
  float* nS   = nM + (size_t)GN * 4;         // N*4
  float* mu   = nS + (size_t)GN * 4;         // 256
  float* var  = mu + 256;                    // 256
  float* en   = var + 256;                   // N
  float* aN   = en + GN;                     // N
  float* bmax = aN + GN;                     // B
  float* bsum = bmax + GB;                   // B
  float* hs   = bsum + GB;                   // B*H
  float* cs   = hs + GB * GH;
  float* rr   = cs + GB * GH;
  float* ga   = rr + GB * GH;
  float* gfb  = ga + GB * GH;
  float* qst  = gfb + GB * GH;               // B*2H
  float* gA   = qst + GB * 2 * GH;           // B*4H
  float* gBuf = gA + GB * 4 * GH;            // B*4H
  float* comb = gBuf + GB * 4 * GH;          // B*4H
  float* ft1  = comb + GB * 4 * GH;          // B*H
  float* ft2  = ft1 + GB * GH;               // B*128
  float* g10  = ft2 + GB * 128;              // B*16 (padded)
  float* tmpB = g10 + GB * 16;               // B*H

  // ---- input embed: h = silu(ln(x @ Wi + bi)) ----
  k_small_gemm<<<GN, GH, 0, stream>>>(x, Wi, bi, b1p, GF, GH);
  k_ln_silu<<<GN, GH, 0, stream>>>(b1p, gi, bei, h, GH);

  // ---- GNN layers ----
  for (int layer = 0; layer < GLAYERS; layer++) {
    if ((layer & 1) == 0) {
      const float *W1=P(pidx+0),*W2=P(pidx+1),*We=P(pidx+2),*pb1=P(pidx+3),
                  *pb2=P(pidx+4),*be1=P(pidx+5),*bee=P(pidx+6),*bn_b=P(pidx+7),
                  *bn_g=P(pidx+8),*g1=P(pidx+9);
      pidx += 10;
      k_fill<<<gup(NH), 256, 0, stream>>>(b1p, 0.f, NH);                       // aggr
      k_edge_aggr<<<GE, GH, 0, stream>>>(h, ea, We, bee, src, dst, b1p);
      k_add<<<gup(NH), 256, 0, stream>>>(h, b1p, b2p, NH);                     // t = h+aggr
      gemm(b2p, W1, pb1, b3p, GN, GH, GH, stream);
      k_ln_silu<<<GN, GH, 0, stream>>>(b3p, g1, be1, b3p, GH);
      gemm(b3p, W2, pb2, b4p, GN, GH, GH, stream);                             // hn
      k_bn_stats<<<GH, 256, 0, stream>>>(b4p, mu, var, GN, GH);
      k_bn_residual<<<gup(NH), 256, 0, stream>>>(h, b4p, mu, var, bn_g, bn_b, NH, GH);
    } else {
      const float *Wbeta=P(pidx+0),*We=P(pidx+1),*Wk=P(pidx+2),*Wq=P(pidx+3),
                  *Wsm=P(pidx+4),*Wv=P(pidx+5),*bee=P(pidx+6),*bk=P(pidx+7),
                  *bn_b=P(pidx+8),*bn_g=P(pidx+9),*bq=P(pidx+10),*bs=P(pidx+11),
                  *bv=P(pidx+12);
      pidx += 13;
      gemm(h, Wq, bq, b1p, GN, GH, GH, stream);                                // Q
      gemm(h, Wk, bk, b2p, GN, GH, GH, stream);                                // K
      gemm(h, Wv, bv, b3p, GN, GH, GH, stream);                                // V
      gemm(h, Wsm, bs, b4p, GN, GH, GH, stream);                               // R
      k_fill<<<gup((size_t)GN * 4), 256, 0, stream>>>(nM, -3.0e38f, (size_t)GN * 4);
      k_fill<<<gup((size_t)GN * 4), 256, 0, stream>>>(nS, 0.f, (size_t)GN * 4);
      k_fill<<<gup(NH), 256, 0, stream>>>(b5p, 0.f, NH);                       // out
      k_attn_logits<<<GE, GH, 0, stream>>>(b1p, b2p, ea, We, bee, src, dst, eL, nM);
      k_attn_expsum<<<gup((size_t)GE * 4), 256, 0, stream>>>(eL, nM, dst, eA, nS, GE);
      k_attn_out<<<GE, GH, 0, stream>>>(b3p, ea, We, bee, src, dst, eA, nS, b5p);
      k_beta<<<GN, GH, 0, stream>>>(b5p, b4p, Wbeta, b1p);                     // hn
      k_bn_stats<<<GH, 256, 0, stream>>>(b1p, mu, var, GN, GH);
      k_bn_residual<<<gup(NH), 256, 0, stream>>>(h, b1p, mu, var, bn_g, bn_b, NH, GH);
    }
  }

  // ---- Set2Set LSTM readout ----
  k_fill<<<gup((size_t)GB * GH), 256, 0, stream>>>(hs, 0.f, (size_t)GB * GH);
  k_fill<<<gup((size_t)GB * GH), 256, 0, stream>>>(cs, 0.f, (size_t)GB * GH);
  k_fill<<<gup((size_t)GB * 2 * GH), 256, 0, stream>>>(qst, 0.f, (size_t)GB * 2 * GH);
  for (int step = 0; step < GSTEPS; step++) {
    gemm(qst, Wih, bih, gA, GB, 2 * GH, 4 * GH, stream);
    gemm(hs, Whh, bhh, gBuf, GB, GH, 4 * GH, stream);
    k_lstm<<<gup((size_t)GB * GH), 256, 0, stream>>>(gA, gBuf, cs, hs);
    k_fill<<<1, GB, 0, stream>>>(bmax, -3.0e38f, GB);
    k_fill<<<1, GB, 0, stream>>>(bsum, 0.f, GB);
    k_fill<<<gup((size_t)GB * GH), 256, 0, stream>>>(rr, 0.f, (size_t)GB * GH);
    k_en<<<GN, GH, 0, stream>>>(h, hs, batch, en, bmax);
    k_segexp<<<gup(GN), 256, 0, stream>>>(en, batch, bmax, aN, bsum, GN);
    k_segwsum<<<GN, GH, 0, stream>>>(aN, bsum, batch, h, rr);
    k_qstar<<<gup((size_t)GB * 2 * GH), 256, 0, stream>>>(hs, rr, qst);
  }

  // ---- gated pooling ----
  gemm(h, Wg1, bg1, b5p, GN, GH, 128, stream);                                 // G1
  k_fill<<<1, GB, 0, stream>>>(bmax, -3.0e38f, GB);
  k_fill<<<1, GB, 0, stream>>>(bsum, 0.f, GB);
  k_fill<<<gup((size_t)GB * GH), 256, 0, stream>>>(ga, 0.f, (size_t)GB * GH);
  k_gate<<<GN, 128, 0, stream>>>(b5p, Wg2, bg2, batch, en, bmax);
  k_segexp<<<gup(GN), 256, 0, stream>>>(en, batch, bmax, aN, bsum, GN);
  k_segwsum<<<GN, GH, 0, stream>>>(aN, bsum, batch, h, ga);

  // ---- global features ----
  k_gfeat<<<1, GB, 0, stream>>>(Om, De, En, ssz, trd, rdd, cen, nA, nB, g10);
  k_small_gemm<<<GB, GH, 0, stream>>>(g10, Wgf1, bgf1, tmpB, 10, GH);
  k_ln_silu<<<GB, GH, 0, stream>>>(tmpB, ggf, begf, tmpB, GH);
  gemm(tmpB, Wgf2, bgf2, gfb, GB, GH, GH, stream);

  // ---- final MLP ----
  k_concat3<<<gup((size_t)GB * 4 * GH), 256, 0, stream>>>(qst, ga, gfb, comb);
  gemm(comb, Wf1, bf1, ft1, GB, 4 * GH, GH, stream);
  k_ln_silu<<<GB, GH, 0, stream>>>(ft1, gf1, bef1, ft1, GH);
  gemm(ft1, Wf2, bf2, ft2, GB, GH, 128, stream);
  k_ln_silu<<<GB, 128, 0, stream>>>(ft2, gf2, bef2, ft2, 128);
  k_final<<<GB, 128, 0, stream>>>(ft2, Wf3, bf3, (float*)d_out);
}
